// YoloBaseModel_70746701300139
// MI455X (gfx1250) — 8-bit matrix operand order repaired and checked by reading, compiled, not run on hardware
//
#include <hip/hip_runtime.h>

typedef __attribute__((ext_vector_type(8))) int v8i;
typedef unsigned long long u64;

#define B_      32
#define N_      10647
#define C_      85
#define NPAD    16384
#define TOPK    1024
#define NCLS    80
#define CONF_T  0.8f
#define NMS_T   0.4f
#define HALF_PAD_X 0.0f
#define HALF_PAD_Y 52.0f
#define SCALE_X (640.0f / 416.0f)
#define SCALE_Y (480.0f / 312.0f)

__global__ void yolo_keys(const float* __restrict__ det, u64* __restrict__ sortbuf) {
    unsigned t = blockIdx.x * blockDim.x + threadIdx.x;
    unsigned b = t / NPAD, n = t % NPAD;
    u64 packed = 0ull;
    if (n < N_) {
        const float* p = det + ((size_t)b * N_ + n) * C_;
        __builtin_prefetch(p, 0, 0);
        float obj = p[4];
        float m = p[5];
        #pragma unroll 4
        for (int c = 1; c < NCLS; ++c) m = fmaxf(m, p[5 + c]);
        float score = obj * m;
        float key = (obj >= CONF_T) ? score : -1.0f;
        unsigned ku = __float_as_uint(key);
        ku = (ku & 0x80000000u) ? ~ku : (ku | 0x80000000u);
        packed = ((u64)ku << 32) | (u64)(0xFFFFFFFFu - n);
    }
    sortbuf[(size_t)b * NPAD + n] = packed;
}

__global__ void yolo_sort(u64* __restrict__ sortbuf) {
    u64* S = sortbuf + (size_t)blockIdx.x * NPAD;
    unsigned tid = threadIdx.x;
    for (unsigned k = 2; k <= NPAD; k <<= 1) {
        for (unsigned j = k >> 1; j > 0; j >>= 1) {
            for (unsigned m = tid; m < NPAD; m += 1024) {
                unsigned p = m ^ j;
                if (p > m) {
                    u64 a = S[m], c = S[p];
                    bool up = ((m & k) == 0);
                    bool sw = up ? (a < c) : (a > c);
                    if (sw) { S[m] = c; S[p] = a; }
                }
            }
            __syncthreads();
        }
    }
}

__global__ void yolo_gather(const float* __restrict__ det, const u64* __restrict__ sortbuf,
                            float4* __restrict__ boxes, float* __restrict__ score,
                            int* __restrict__ cid, int* __restrict__ valid) {
    unsigned t = blockIdx.x * blockDim.x + threadIdx.x;
    unsigned b = t >> 10, r = t & (TOPK - 1);
    u64 packed = sortbuf[(size_t)b * NPAD + r];
    unsigned n = 0xFFFFFFFFu - (unsigned)(packed & 0xFFFFFFFFu);
    if (n >= N_) n = 0;
    const float* p = det + ((size_t)b * N_ + n) * C_;
    float cx = p[0], cy = p[1], w = p[2], h = p[3], obj = p[4];
    float best = p[5]; int bi = 0;
    for (int c = 1; c < NCLS; ++c) {
        float v = p[5 + c];
        if (v > best) { best = v; bi = c; }
    }
    boxes[t] = make_float4(cx - 0.5f * w, cy - 0.5f * h, cx + 0.5f * w, cy + 0.5f * h);
    score[t] = obj * best;
    cid[t]   = bi;
    valid[t] = (obj >= CONF_T) ? 1 : 0;
}

__global__ void yolo_supmask(const float4* __restrict__ boxes, const int* __restrict__ cid,
                             unsigned short* __restrict__ mask) {
    unsigned lane  = threadIdx.x & 31u;
    unsigned wave  = threadIdx.x >> 5;
    unsigned tile  = blockIdx.x * 8u + wave;
    unsigned b     = tile >> 12;
    unsigned ti    = (tile >> 6) & 63u;
    unsigned tj    = tile & 63u;
    unsigned i0    = ti * 16u, j0 = tj * 16u;
    unsigned base  = b * TOPK;

    int rowc = cid[base + i0 + (lane & 15u)];
    int colc = cid[base + j0 + (lane & 15u)];

    int rbyte = (int)(1u << (8 * (rowc & 3)));
    int cbyte = (int)(1u << (8 * (colc & 3)));
    int rchunk = rowc >> 2;
    int cchunk = colc >> 2;
    v8i a0 = {}, a1 = {}, b0 = {}, b1 = {};
    #pragma unroll
    for (int v = 0; v < 8; ++v) {
        int achunk = ((v >> 2) * 8) + (((v >> 1) & 1) * 4) + (v & 1) + ((lane >= 16) ? 2 : 0);
        a0[v] = (rchunk == achunk)      ? rbyte : 0;
        a1[v] = (rchunk == achunk + 16) ? rbyte : 0;
        int bchunk = ((v >> 2) * 8) + (((v >> 1) & 1) * 4) + (v & 1) + ((lane >= 16) ? 2 : 0);
        b0[v] = (cchunk == bchunk)      ? cbyte : 0;
        b1[v] = (cchunk == bchunk + 16) ? cbyte : 0;
    }
    v8i acc0 = {}, acc1 = {};
    acc0 = __builtin_amdgcn_wmma_i32_16x16x64_iu8(false, a0, false, b0, acc0, false, false);
    acc1 = __builtin_amdgcn_wmma_i32_16x16x64_iu8(false, a1, false, b1, acc1, false, false);

    float4 cb = boxes[base + j0 + (lane & 15u)];
    float areaC = fmaxf(cb.z - cb.x, 0.f) * fmaxf(cb.w - cb.y, 0.f);
    unsigned gj = j0 + (lane & 15u);

    #pragma unroll
    for (int v = 0; v < 8; ++v) {
        unsigned M  = (unsigned)v + ((lane >= 16) ? 8u : 0u);
        unsigned gi = i0 + M;
        float4 rb = boxes[base + gi];
        float areaR = fmaxf(rb.z - rb.x, 0.f) * fmaxf(rb.w - rb.y, 0.f);
        float ix1 = fmaxf(rb.x, cb.x), iy1 = fmaxf(rb.y, cb.y);
        float ix2 = fminf(rb.z, cb.z), iy2 = fminf(rb.w, cb.w);
        float inter = fmaxf(ix2 - ix1, 0.f) * fmaxf(iy2 - iy1, 0.f);
        float unionEps = areaR + areaC - inter + 1e-16f;
        bool pred = ((acc0[v] | acc1[v]) != 0) && (inter > NMS_T * unionEps) && (gj > gi);
        unsigned bal = __builtin_amdgcn_ballot_w32(pred);
        if (lane == 0) {
            mask[((size_t)base + i0 + v) * 64 + tj]     = (unsigned short)(bal & 0xFFFFu);
            mask[((size_t)base + i0 + v + 8) * 64 + tj] = (unsigned short)(bal >> 16);
        }
    }
}

__global__ void yolo_scan(const float4* __restrict__ boxes, const float* __restrict__ score,
                          const int* __restrict__ cid, const int* __restrict__ valid,
                          const unsigned short* __restrict__ mask, float* __restrict__ out) {
    __shared__ int validS[TOPK];
    __shared__ unsigned aliveW[64];
    __shared__ unsigned char keep[TOPK];
    __shared__ int keepFlag;
    unsigned b = blockIdx.x, tid = threadIdx.x;
    unsigned base = b * TOPK;

    {
        unsigned ldsOff = (unsigned)(size_t)(&validS[0]) + tid * 16u;
        u64 gaddr = (u64)(const void*)(valid + base) + (u64)tid * 16u;
        asm volatile("global_load_async_to_lds_b128 %0, %1, off"
                     :: "v"(ldsOff), "v"(gaddr) : "memory");
        asm volatile("s_wait_asynccnt 0x0" ::: "memory");
    }
    if (tid < 64) aliveW[tid] = 0xFFFFu;
    __syncthreads();

    for (int i = 0; i < TOPK; ++i) {
        if (tid < 64 && i + 1 < TOPK)
            __builtin_prefetch(&mask[((size_t)base + i + 1) * 64 + tid], 0, 0);
        if (tid == 0) {
            int k = ((aliveW[i >> 4] >> (i & 15)) & 1u) && validS[i];
            keepFlag = k;
            keep[i] = (unsigned char)k;
        }
        __syncthreads();
        if (keepFlag && tid < 64)
            aliveW[tid] &= ~(unsigned)mask[((size_t)base + i) * 64 + tid];
        __syncthreads();
    }

    for (unsigned r = tid; r < TOPK; r += blockDim.x) {
        float k = keep[r] ? 1.0f : 0.0f;
        float4 bx = boxes[base + r];
        float x1 = (bx.x - HALF_PAD_X) * SCALE_X;
        float y1 = (bx.y - HALF_PAD_Y) * SCALE_Y;
        float x2 = (bx.z - HALF_PAD_X) * SCALE_X;
        float y2 = (bx.w - HALF_PAD_Y) * SCALE_Y;
        float* o = out + ((size_t)base + r) * 7;
        o[0] = x1 * k; o[1] = y1 * k; o[2] = x2 * k; o[3] = y2 * k;
        o[4] = score[base + r] * k;
        o[5] = (float)cid[base + r];
        o[6] = k;
    }
}

extern "C" void kernel_launch(void* const* d_in, const int* in_sizes, int n_in,
                              void* d_out, int out_size, void* d_ws, size_t ws_size,
                              hipStream_t stream) {
    (void)in_sizes; (void)n_in; (void)out_size; (void)ws_size;
    const float* det = (const float*)d_in[0];
    float* out = (float*)d_out;
    char* ws = (char*)d_ws;

    u64*            sortbuf = (u64*)            (ws + 0);
    float4*         boxes   = (float4*)         (ws + 4194304);
    float*          score   = (float*)          (ws + 4718592);
    int*            cid     = (int*)            (ws + 4849664);
    int*            valid   = (int*)            (ws + 4980736);
    unsigned short* mask    = (unsigned short*) (ws + 5111808);

    yolo_keys   <<<(B_ * NPAD) / 256, 256, 0, stream>>>(det, sortbuf);
    yolo_sort   <<<B_, 1024, 0, stream>>>(sortbuf);
    yolo_gather <<<(B_ * TOPK) / 256, 256, 0, stream>>>(det, sortbuf, boxes, score, cid, valid);
    yolo_supmask<<<(B_ * 64 * 64) / 8, 256, 0, stream>>>(boxes, cid, mask);
    yolo_scan   <<<B_, 256, 0, stream>>>(boxes, score, cid, valid, mask, out);
}
